// HeteroGNN_61065845015098
// MI455X (gfx1250) — matrix operand order repaired and checked by reading, not compiled, not run on hardware
//
#include <hip/hip_runtime.h>

// ============================================================================
// HeteroGNN (2-layer GAT + pair MLP) for MI455X / gfx1250, wave32.
//
// Roofline: ~105 GFLOP of GEMM (WMMA f16->f32) vs ~6 GB of HBM traffic
// (edge gathers + 128-wide float atomics dominate) => memory-bound at
// 23.3 TB/s (~0.26 ms floor). All dense math uses v_wmma_f32_16x16x32_f16;
// edge_emb concat is fused into the MLP-1 WMMA A-fragment build to avoid a
// 410 MB intermediate.
// ============================================================================

#define HDIM 128
#define NEG_SLOPE 0.2f
#define BN_EPS 1e-5f

typedef __attribute__((ext_vector_type(16))) _Float16 v16h;
typedef __attribute__((ext_vector_type(8)))  float    v8f;

// |x| via raw v_and_b32: opaque to the SLP vectorizer / ISel fabs pattern,
// which miscompiles (backend crash) when fabs is formed over vectorized
// extracts on this toolchain. Same cost as fabsf.
__device__ __forceinline__ float fabs_bits(float x) {
  float r;
  asm("v_and_b32 %0, 0x7fffffff, %1" : "=v"(r) : "v"(x));
  return r;
}

// ---------------------------------------------------------------- fills ----
__global__ void fill_f32_kernel(float* __restrict__ p, float v, int n) {
  int i = blockIdx.x * blockDim.x + threadIdx.x;
  if (i < n) p[i] = v;
}
__global__ void fill_u32_kernel(unsigned* __restrict__ p, unsigned v, int n) {
  int i = blockIdx.x * blockDim.x + threadIdx.x;
  if (i < n) p[i] = v;
}

// ------------------------------------------------- ordered-float helpers ----
__device__ __forceinline__ unsigned ord_enc(float f) {
  unsigned u = __float_as_uint(f);
  return (u & 0x80000000u) ? ~u : (u | 0x80000000u);
}
__device__ __forceinline__ float ord_dec(unsigned u) {
  return (u & 0x80000000u) ? __uint_as_float(u & 0x7fffffffu)
                           : __uint_as_float(~u);
}

// ------------------------------------------------------------ WMMA GEMM ----
// C[M,N] = A[M,K] @ B[K,N] (+bias)(+relu). f32 operands converted inline to
// f16, f32 accumulate. One wave = one 16x16 tile; 8 waves/block tile N.
// Requires M%16==0, K%32==0, N%16==0 (all call sites satisfy this exactly).
// Fragment layouts per CDNA5 ISA 7.12.2 (16-bit A 16x32; B same K order
// as A; C/D f32 8 VGPRs).
__global__ __launch_bounds__(256)
void gemm_wmma(const float* __restrict__ A, const float* __restrict__ B,
               const float* __restrict__ bias, float* __restrict__ C,
               int M, int N, int K, int flags) {  // flags: 1=bias, 2=relu
  const int wave  = threadIdx.x >> 5;
  const int lane  = threadIdx.x & 31;
  const int tileN = blockIdx.y * 8 + wave;
  if (tileN * 16 >= N) return;               // wave-uniform: EXEC stays all-1s
  const int row0 = blockIdx.x << 4;
  const int col0 = tileN << 4;
  const int h = lane >> 4;                   // half-wave
  const int r = lane & 15;
  v8f acc = {};
  for (int k0 = 0; k0 < K; k0 += 32) {
    v16h af, bf;
    const float* arow = A + (size_t)(row0 + r) * K + k0;
#pragma unroll
    for (int v = 0; v < 8; ++v) {
      int kb = ((v >> 2) << 4) + (h << 3) + ((v & 3) << 1);   // A 16x32 layout
      af[2 * v]     = (_Float16)arow[kb];
      af[2 * v + 1] = (_Float16)arow[kb + 1];
    }
    const float* bcol = B + (size_t)(k0 + (h << 3)) * N + col0 + r;
#pragma unroll
    for (int v = 0; v < 8; ++v) {            // B: same K order as A 16x32 layout
      int kk = ((v >> 2) << 4) + ((v & 3) << 1);
      bf[2 * v]     = (_Float16)bcol[(size_t)kk * N];
      bf[2 * v + 1] = (_Float16)bcol[(size_t)(kk + 1) * N];
    }
    acc = __builtin_amdgcn_wmma_f32_16x16x32_f16(false, af, false, bf,
                                                 (short)0, acc, false, false);
  }
#pragma unroll
  for (int rr = 0; rr < 8; ++rr) {
    int n = col0 + r;
    float v = acc[rr];
    if (flags & 1) v += bias[n];
    if (flags & 2) v = fmaxf(v, 0.f);
    C[(size_t)(row0 + rr + (h << 3)) * N + n] = v;
  }
}

// --------------------------------------- fused pair-MLP layer 1 (WMMA) ----
// H[P,256] = relu(concat(m, p, m*p, |m-p|)[P,512] @ W1[512,256] + b1), with
// the gather/concat fused into the A-fragment build (saves 410 MB traffic).
__global__ __launch_bounds__(256)
void mlp1_fused_wmma(const float* __restrict__ meta, const float* __restrict__ pro,
                     const int* __restrict__ midx, const int* __restrict__ pidx,
                     const float* __restrict__ W, const float* __restrict__ bias,
                     float* __restrict__ Hout, int P, int N) {
  const int wave  = threadIdx.x >> 5;
  const int lane  = threadIdx.x & 31;
  const int tileN = blockIdx.y * 8 + wave;
  if (tileN * 16 >= N) return;
  const int row0 = blockIdx.x << 4;
  const int col0 = tileN << 4;
  const int h = lane >> 4, r = lane & 15;
  const float* mrow = meta + (size_t)midx[row0 + r] * HDIM;
  const float* prow = pro  + (size_t)pidx[row0 + r] * HDIM;
  v8f acc = {};
  for (int k0 = 0; k0 < 4 * HDIM; k0 += 32) {
    const int seg = k0 >> 7;                 // 32 | 128 => chunk in one segment
    v16h af, bf;
#pragma unroll
    for (int v = 0; v < 8; ++v) {
      int kb = k0 + ((v >> 2) << 4) + (h << 3) + ((v & 3) << 1);
      int k_a = kb & (HDIM - 1), k_b = (kb + 1) & (HDIM - 1);
      float m0 = mrow[k_a], q0 = prow[k_a];
      float m1 = mrow[k_b], q1 = prow[k_b];
      float e0 = seg == 0 ? m0 : seg == 1 ? q0 : seg == 2 ? m0 * q0
                                                          : fabs_bits(m0 - q0);
      float e1 = seg == 0 ? m1 : seg == 1 ? q1 : seg == 2 ? m1 * q1
                                                          : fabs_bits(m1 - q1);
      af[2 * v]     = (_Float16)e0;
      af[2 * v + 1] = (_Float16)e1;
    }
    const float* bcol = W + (size_t)(k0 + (h << 3)) * N + col0 + r;
#pragma unroll
    for (int v = 0; v < 8; ++v) {
      int kk = ((v >> 2) << 4) + ((v & 3) << 1);
      bf[2 * v]     = (_Float16)bcol[(size_t)kk * N];
      bf[2 * v + 1] = (_Float16)bcol[(size_t)(kk + 1) * N];
    }
    acc = __builtin_amdgcn_wmma_f32_16x16x32_f16(false, af, false, bf,
                                                 (short)0, acc, false, false);
  }
#pragma unroll
  for (int rr = 0; rr < 8; ++rr) {
    float v = fmaxf(acc[rr] + bias[col0 + r], 0.f);
    Hout[(size_t)(row0 + rr + (h << 3)) * N + col0 + r] = v;
  }
}

// ---------------------------------------------------- attention scalars ----
// out[row] = dot(X[row, 0:128], vec). One wave per row, shfl_xor reduce.
__global__ void rowdot_kernel(const float* __restrict__ X,
                              const float* __restrict__ vec,
                              float* __restrict__ out, int N) {
  int row  = (blockIdx.x * blockDim.x + threadIdx.x) >> 5;
  int lane = threadIdx.x & 31;
  if (row >= N) return;
  const float* xp = X + (size_t)row * HDIM;
  float s = 0.f;
#pragma unroll
  for (int c = lane; c < HDIM; c += 32) s += xp[c] * vec[c];
#pragma unroll
  for (int off = 16; off; off >>= 1) s += __shfl_xor(s, off, 32);
  if (lane == 0) out[row] = s;
}

// ----------------------------------------------------------- edge passes ----
__global__ void edge_logits_kernel(const int* __restrict__ ei,
                                   const float* __restrict__ as,
                                   const float* __restrict__ ad,
                                   float* __restrict__ ebuf,
                                   unsigned* __restrict__ emax, int E) {
  int i = blockIdx.x * blockDim.x + threadIdx.x;
  if (i >= E) return;
  int s = ei[i], d = ei[E + i];
  float e = as[s] + ad[d];
  e = e > 0.f ? e : NEG_SLOPE * e;
  ebuf[i] = e;
  atomicMax(&emax[d], ord_enc(e));
}

__global__ void edge_exp_kernel(const int* __restrict__ ei,
                                float* __restrict__ ebuf,
                                const unsigned* __restrict__ emax,
                                float* __restrict__ denom, int E) {
  int i = blockIdx.x * blockDim.x + threadIdx.x;
  if (i >= E) return;
  int d = ei[E + i];
  float ex = __expf(ebuf[i] - ord_dec(emax[d]));
  ebuf[i] = ex;
  atomicAdd(&denom[d], ex);
}

// One wave per edge; lane c covers channels c, c+32, c+64, c+96 (coalesced).
__global__ void edge_scatter_kernel(const int* __restrict__ ei,
                                    const float* __restrict__ ebuf,
                                    const float* __restrict__ denom,
                                    const float* __restrict__ hs,
                                    float* __restrict__ out, int E) {
  int eidx = (blockIdx.x * blockDim.x + threadIdx.x) >> 5;
  int lane = threadIdx.x & 31;
  if (eidx >= E) return;
  int s = ei[eidx], d = ei[E + eidx];
  float w = ebuf[eidx] / denom[d];
  const float* hp = hs + (size_t)s * HDIM;
  float* op = out + (size_t)d * HDIM;
#pragma unroll
  for (int c = lane; c < HDIM; c += 32) atomicAdd(&op[c], w * hp[c]);
}

// ------------------------------- mean(2 rel)+bias -> relu -> BN stats ------
// Thread = channel (blockDim=128); block covers a row range, 2 atomics/thread.
__global__ __launch_bounds__(HDIM)
void relu_combine_stats_kernel(float* __restrict__ X,
                               const float* __restrict__ b1,
                               const float* __restrict__ b2,
                               float* __restrict__ stats,  // [256] sum|sumsq
                               int N, int rows_per_block) {
  int c  = threadIdx.x;
  int r0 = blockIdx.x * rows_per_block;
  int r1 = min(r0 + rows_per_block, N);
  float bb = b1[c] + b2[c];
  float s = 0.f, s2 = 0.f;
  for (int r = r0; r < r1; ++r) {
    float v = fmaxf(0.5f * (X[(size_t)r * HDIM + c] + bb), 0.f);
    X[(size_t)r * HDIM + c] = v;
    s += v; s2 += v * v;
  }
  atomicAdd(&stats[c], s);
  atomicAdd(&stats[HDIM + c], s2);
}

__global__ void bn_apply_kernel(float* __restrict__ X,
                                const float* __restrict__ stats,
                                const float* __restrict__ g,
                                const float* __restrict__ b, int N) {
  int i = blockIdx.x * blockDim.x + threadIdx.x;
  if (i >= N * HDIM) return;
  int c = i & (HDIM - 1);
  float inv_n = 1.f / (float)N;
  float mean  = stats[c] * inv_n;
  float var   = stats[HDIM + c] * inv_n - mean * mean;
  X[i] = g[c] * (X[i] - mean) * rsqrtf(var + BN_EPS) + b[c];
}

// ------------------------------------------------------------ MLP layer 2 --
__global__ void mlp2_kernel(const float* __restrict__ Hm,
                            const float* __restrict__ w2,
                            const float* __restrict__ b2,
                            float* __restrict__ out, int P) {
  int row  = (blockIdx.x * blockDim.x + threadIdx.x) >> 5;
  int lane = threadIdx.x & 31;
  if (row >= P) return;
  const float* hp = Hm + (size_t)row * (2 * HDIM);
  float s = 0.f;
#pragma unroll
  for (int c = lane; c < 2 * HDIM; c += 32) s += hp[c] * w2[c];
#pragma unroll
  for (int off = 16; off; off >>= 1) s += __shfl_xor(s, off, 32);
  if (lane == 0) out[row] = s + b2[0];
}

// ============================================================================
// Host orchestration. Params assumed flattened in setup_inputs() dict
// insertion order (recursively):
//   0 x_metabolite  1 x_protein  2 ei_pp  3 ei_mm  4 ei_mp  5 ei_pm
//   6 metabolite_idx  7 protein_idx
//   8 proj_m_w  9 proj_m_b  10 proj_p_w  11 proj_p_b
//   layer L base = 12 + 24*L:
//     +0..4  pp {w_src,w_dst,a_src,a_dst,bias}
//     +5..9  mm   +10..14 mp   +15..19 pm
//     +20 bn_p_g  +21 bn_p_b  +22 bn_m_g  +23 bn_m_b
//   60 mlp_w1  61 mlp_b1  62 mlp_w2  63 mlp_b2
// ============================================================================
extern "C" void kernel_launch(void* const* d_in, const int* in_sizes, int n_in,
                              void* d_out, int out_size, void* d_ws, size_t ws_size,
                              hipStream_t stream) {
  auto F = [&](int i) { return (const float*)d_in[i]; };
  auto I = [&](int i) { return (const int*)d_in[i]; };
  auto cdiv = [](int a, int b) { return (a + b - 1) / b; };

  const int Km = in_sizes[8] / HDIM;        // 768
  const int Kp = in_sizes[10] / HDIM;       // 1280
  const int Nm = in_sizes[0] / Km;          // 50000
  const int Np = in_sizes[1] / Kp;          // 50000
  const int E  = in_sizes[2] / 2;           // 500000
  const int P  = in_sizes[6];               // 200000
  const int Nmax = Nm > Np ? Nm : Np;

  float* ws = (float*)d_ws;
  size_t o = 0;
  auto alloc = [&](size_t n) { float* p = ws + o; o += (n + 63) & ~(size_t)63; return p; };
  float*    meta  = alloc((size_t)Nm * HDIM);
  float*    pro   = alloc((size_t)Np * HDIM);
  float*    nmeta = alloc((size_t)Nm * HDIM);
  float*    npro  = alloc((size_t)Np * HDIM);
  float*    hs    = alloc((size_t)Nmax * HDIM);
  float*    hd    = alloc((size_t)Nmax * HDIM);
  float*    a_s   = alloc(Nmax);
  float*    a_d   = alloc(Nmax);
  unsigned* emax  = (unsigned*)alloc(Nmax);
  float*    denom = alloc(Nmax);
  float*    ebuf  = alloc(E);
  float*    stats = alloc(2 * HDIM);
  float*    hbuf  = alloc((size_t)P * 2 * HDIM);

  const dim3 blk(256);

  // Input projections (WMMA, bias).
  gemm_wmma<<<dim3(Nm / 16, 1), blk, 0, stream>>>(F(0), F(8),  F(9),  meta, Nm, HDIM, Km, 1);
  gemm_wmma<<<dim3(Np / 16, 1), blk, 0, stream>>>(F(1), F(10), F(11), pro,  Np, HDIM, Kp, 1);

  struct Rel { int pidx; const float* src; int nsrc; const int* ei; };
  for (int L = 0; L < 2; ++L) {
    const int base = 12 + L * 24;
    for (int side = 0; side < 2; ++side) {  // 0: dst=pro (pp,mp)  1: dst=meta (mm,pm)
      float*       dstbuf = side == 0 ? npro : nmeta;
      const float* dstemb = side == 0 ? pro  : meta;
      const int    ndst   = side == 0 ? Np   : Nm;
      Rel rels[2];
      if (side == 0) { rels[0] = {base + 0,  pro,  Np, I(2)};    // pp
                       rels[1] = {base + 10, meta, Nm, I(4)}; }  // mp
      else           { rels[0] = {base + 5,  meta, Nm, I(3)};    // mm
                       rels[1] = {base + 15, pro,  Np, I(5)}; }  // pm

      fill_f32_kernel<<<cdiv(ndst * HDIM, 256), blk, 0, stream>>>(dstbuf, 0.f, ndst * HDIM);
      for (int q = 0; q < 2; ++q) {
        const Rel& rl = rels[q];
        gemm_wmma<<<dim3(rl.nsrc / 16, 1), blk, 0, stream>>>(rl.src, F(rl.pidx + 0), nullptr, hs, rl.nsrc, HDIM, HDIM, 0);
        gemm_wmma<<<dim3(ndst / 16, 1),    blk, 0, stream>>>(dstemb, F(rl.pidx + 1), nullptr, hd, ndst,    HDIM, HDIM, 0);
        rowdot_kernel<<<cdiv(rl.nsrc, 8), blk, 0, stream>>>(hs, F(rl.pidx + 2), a_s, rl.nsrc);
        rowdot_kernel<<<cdiv(ndst, 8),    blk, 0, stream>>>(hd, F(rl.pidx + 3), a_d, ndst);
        fill_u32_kernel<<<cdiv(ndst, 256), blk, 0, stream>>>(emax, 0u, ndst);
        fill_f32_kernel<<<cdiv(ndst, 256), blk, 0, stream>>>(denom, 0.f, ndst);
        edge_logits_kernel <<<cdiv(E, 256), blk, 0, stream>>>(rl.ei, a_s, a_d, ebuf, emax, E);
        edge_exp_kernel    <<<cdiv(E, 256), blk, 0, stream>>>(rl.ei, ebuf, emax, denom, E);
        edge_scatter_kernel<<<cdiv(E, 8),   blk, 0, stream>>>(rl.ei, ebuf, denom, hs, dstbuf, E);
      }
      const int bn = base + 20 + (side == 0 ? 0 : 2);
      fill_f32_kernel<<<1, blk, 0, stream>>>(stats, 0.f, 2 * HDIM);
      relu_combine_stats_kernel<<<cdiv(ndst, 32), dim3(HDIM), 0, stream>>>(
          dstbuf, F(rels[0].pidx + 4), F(rels[1].pidx + 4), stats, ndst, 32);
      bn_apply_kernel<<<cdiv(ndst * HDIM, 256), blk, 0, stream>>>(
          dstbuf, stats, F(bn), F(bn + 1), ndst);
    }
    { float* t = meta; meta = nmeta; nmeta = t; }
    { float* t = pro;  pro  = npro;  npro  = t; }
  }

  // Pair scoring MLP: fused gather/concat GEMM (WMMA) then 256->1 dot.
  mlp1_fused_wmma<<<dim3(P / 16, 2), blk, 0, stream>>>(
      meta, pro, I(6), I(7), F(60), F(61), hbuf, P, 2 * HDIM);
  mlp2_kernel<<<cdiv(P, 8), blk, 0, stream>>>(hbuf, F(62), F(63), (float*)d_out, P);

  (void)n_in; (void)out_size; (void)ws_size;
}
